// GFootballMamba_59828894433549
// MI455X (gfx1250) — matrix operand order repaired and checked by reading, not compiled, not run on hardware
//
#include <hip/hip_runtime.h>

// ---------------------------------------------------------------------------
// GFootball Mamba forward on gfx1250 (MI455X). Compute path: f16 WMMA GEMMs
// (v_wmma_f32_16x16x32_f16) with 2x2 register tiling (32x32 per wave, 4 WMMAs
// per 6 fragment loads), VALU kernels for LN / conv / selective-scan.
// All intermediates in d_ws; deterministic, capture-safe.
// ---------------------------------------------------------------------------

typedef _Float16 f16;
typedef __attribute__((ext_vector_type(16))) _Float16 v16h;
typedef __attribute__((ext_vector_type(8)))  _Float16 v8h;
typedef __attribute__((ext_vector_type(8)))  float    v8f;

#define B_SZ     1024
#define NFRAMES  4
#define FRAME    115
#define DMODEL   256
#define NLAYERS  4
#define DSTATE   16
#define DCONV    4
#define DINNER   512
#define MROWS    (B_SZ * NFRAMES)   // 4096

#define EPI_NONE     0
#define EPI_SOFTPLUS 1
#define EPI_MISH     2

// ------------------------------ device helpers ------------------------------

__device__ __forceinline__ float softplus_f(float x) {
    return (x > 20.f) ? x : log1pf(__expf(x));
}
__device__ __forceinline__ float mish_f(float x) {
    return x * tanhf(softplus_f(x));
}
__device__ __forceinline__ float silu_f(float x) {
    return x * (1.f / (1.f + __expf(-x)));
}
__device__ __forceinline__ float wave_sum(float v) {
    #pragma unroll
    for (int off = 16; off > 0; off >>= 1) v += __shfl_xor(v, off, 32);
    return v;
}

__device__ __forceinline__ v8f wmma_f16(v16h a, v16h b, v8f c) {
    return __builtin_amdgcn_wmma_f32_16x16x32_f16(false, a, false, b,
                                                  (short)0, c, false, false);
}

// ------------------------------ WMMA GEMM -----------------------------------
// C[M,N] = act( A[M,Kp](f16) * Bt[Npad,Kp]^T(f16) + bias + residual )
// One wave computes a 32x32 tile (2x2 of 16x16), 4 waves / block.
// Requirements: Kp % 32 == 0, M % 32 == 0, Bt padded+zeroed to Npad % 32 == 0.

__global__ void gemm_wmma_kernel(const f16* __restrict__ A,
                                 const f16* __restrict__ Bt,
                                 const float* __restrict__ bias,
                                 float* __restrict__ C,
                                 f16* __restrict__ C16,
                                 const float* __restrict__ residual,
                                 int M, int N, int Kp, int ldc, int epi,
                                 int tilesN, int totalTiles) {
    const int wid  = threadIdx.x >> 5;
    const int lane = threadIdx.x & 31;
    const int tile = blockIdx.x * (blockDim.x >> 5) + wid;
    if (tile >= totalTiles) return;              // wave-uniform exit (EXEC stays all-ones)

    const int tm = tile / tilesN;
    const int tn = tile % tilesN;
    const int h  = lane >> 4;                    // half-wave id
    const int r0 = lane & 15;

    const f16* Arow0 = A  + (size_t)(tm * 32 + r0) * Kp;
    const f16* Arow1 = Arow0 + (size_t)16 * Kp;
    const f16* Brow0 = Bt + (size_t)(tn * 32 + r0) * Kp;
    const f16* Brow1 = Brow0 + (size_t)16 * Kp;

    v8f acc00 = {}, acc01 = {}, acc10 = {}, acc11 = {};
    for (int kk = 0; kk < Kp; kk += 32) {
        // A fragments: lane row r0; half-wave h holds K-subrange [16h..16h+15]
        v8h a0lo = *(const v8h*)(Arow0 + kk + 16 * h);
        v8h a0hi = *(const v8h*)(Arow0 + kk + 16 * h + 8);
        v8h a1lo = *(const v8h*)(Arow1 + kk + 16 * h);
        v8h a1hi = *(const v8h*)(Arow1 + kk + 16 * h + 8);
        v16h a0, a1;
        #pragma unroll
        for (int i = 0; i < 8; ++i) {
            a0[i] = a0lo[i]; a0[i + 8] = a0hi[i];
            a1[i] = a1lo[i]; a1[i + 8] = a1hi[i];
        }
        // B fragments: lane col r0; half-wave h holds K-subrange [16h..16h+15]
        v16h b0 = *(const v16h*)(Brow0 + kk + 16 * h);
        v16h b1 = *(const v16h*)(Brow1 + kk + 16 * h);
        acc00 = wmma_f16(a0, b0, acc00);
        acc01 = wmma_f16(a0, b1, acc01);
        acc10 = wmma_f16(a1, b0, acc10);
        acc11 = wmma_f16(a1, b1, acc11);
    }

    #pragma unroll
    for (int sm = 0; sm < 2; ++sm) {
        #pragma unroll
        for (int sn = 0; sn < 2; ++sn) {
            const v8f acc = (sm == 0) ? ((sn == 0) ? acc00 : acc01)
                                      : ((sn == 0) ? acc10 : acc11);
            const int n   = tn * 32 + sn * 16 + r0;
            const bool ok = (n < N);
            const float bb = (bias && ok) ? bias[n] : 0.f;
            #pragma unroll
            for (int r = 0; r < 8; ++r) {
                const int m = tm * 32 + sm * 16 + h * 8 + r;  // C/D: vgpr r -> row r + 8h
                float v = acc[r] + bb;
                if (residual && ok) v += residual[(size_t)m * N + n];
                if (epi == EPI_SOFTPLUS)      v = softplus_f(v);
                else if (epi == EPI_MISH)     v = mish_f(v);
                if (ok) {
                    C[(size_t)m * ldc + n] = v;
                    if (C16) C16[(size_t)m * N + n] = (f16)v;
                }
            }
        }
    }
}

// --------------------- weight convert/transpose (f32 -> f16^T) ---------------
// Wt[n, k] = W[k, n], zero-padded to [Npad, Kpad].

__global__ void wcvt_kernel(const float* __restrict__ W, f16* __restrict__ Wt,
                            int K, int N, int Kp, int Np) {
    int idx = blockIdx.x * blockDim.x + threadIdx.x;
    if (idx >= Kp * Np) return;
    int n = idx / Kp, k = idx % Kp;
    f16 v = (f16)0.f;
    if (k < K && n < N) v = (f16)W[(size_t)k * N + n];
    Wt[idx] = v;
}

// ---------------------------- obs -> padded f16 ------------------------------

__global__ void obs_pad_kernel(const float* __restrict__ obs, f16* __restrict__ A0) {
    int idx = blockIdx.x * blockDim.x + threadIdx.x;      // MROWS * 128
    if (idx >= MROWS * 128) return;
    int r = idx >> 7, k = idx & 127;
    int b = r >> 2, l = r & 3;
    A0[idx] = (k < FRAME) ? (f16)obs[(size_t)b * (NFRAMES * FRAME) + l * FRAME + k]
                          : (f16)0.f;
}

// ---------------- LN variants: wave-per-row over 256 cols --------------------

__global__ void embed_post_kernel(const float* __restrict__ e,
                                  const float* __restrict__ g,
                                  const float* __restrict__ be,
                                  const float* __restrict__ pos,
                                  float* __restrict__ x) {
    int row = blockIdx.x * 8 + (threadIdx.x >> 5);
    int lane = threadIdx.x & 31;
    const float* er = e + (size_t)row * DMODEL;
    float v[8], s = 0.f, s2 = 0.f;
    #pragma unroll
    for (int j = 0; j < 8; ++j) {
        v[j] = er[lane + 32 * j];
        s += v[j]; s2 += v[j] * v[j];
    }
    s = wave_sum(s); s2 = wave_sum(s2);
    float mean = s * (1.f / DMODEL);
    float var  = s2 * (1.f / DMODEL) - mean * mean;
    float rs   = rsqrtf(var + 1e-5f);
    int l = row & 3;
    #pragma unroll
    for (int j = 0; j < 8; ++j) {
        int c = lane + 32 * j;
        float t = (v[j] - mean) * rs * g[c] + be[c];
        x[(size_t)row * DMODEL + c] = mish_f(t) + pos[l * DMODEL + c];
    }
}

__global__ void preln_kernel(const float* __restrict__ x,
                             const float* __restrict__ g,
                             const float* __restrict__ be,
                             f16* __restrict__ h16) {
    int row = blockIdx.x * 8 + (threadIdx.x >> 5);
    int lane = threadIdx.x & 31;
    const float* xr = x + (size_t)row * DMODEL;
    float v[8], s = 0.f, s2 = 0.f;
    #pragma unroll
    for (int j = 0; j < 8; ++j) {
        v[j] = xr[lane + 32 * j];
        s += v[j]; s2 += v[j] * v[j];
    }
    s = wave_sum(s); s2 = wave_sum(s2);
    float mean = s * (1.f / DMODEL);
    float rs   = rsqrtf(s2 * (1.f / DMODEL) - mean * mean + 1e-5f);
    #pragma unroll
    for (int j = 0; j < 8; ++j) {
        int c = lane + 32 * j;
        h16[(size_t)row * DMODEL + c] = (f16)((v[j] - mean) * rs * g[c] + be[c]);
    }
}

__global__ void pool_ln_kernel(const float* __restrict__ x,
                               const float* __restrict__ g,
                               const float* __restrict__ be,
                               f16* __restrict__ o16) {
    int b = blockIdx.x * 8 + (threadIdx.x >> 5);
    int lane = threadIdx.x & 31;
    float v[8], s = 0.f, s2 = 0.f;
    #pragma unroll
    for (int j = 0; j < 8; ++j) {
        int c = lane + 32 * j;
        float t = 0.25f * (x[(size_t)(b * 4 + 0) * DMODEL + c] +
                           x[(size_t)(b * 4 + 1) * DMODEL + c] +
                           x[(size_t)(b * 4 + 2) * DMODEL + c] +
                           x[(size_t)(b * 4 + 3) * DMODEL + c]);
        v[j] = t; s += t; s2 += t * t;
    }
    s = wave_sum(s); s2 = wave_sum(s2);
    float mean = s * (1.f / DMODEL);
    float rs   = rsqrtf(s2 * (1.f / DMODEL) - mean * mean + 1e-5f);
    #pragma unroll
    for (int j = 0; j < 8; ++j) {
        int c = lane + 32 * j;
        o16[(size_t)b * DMODEL + c] = (f16)((v[j] - mean) * rs * g[c] + be[c]);
    }
}

// -------------------- causal depthwise conv (K=4) + SiLU ---------------------
// inproj rows are [B*L, 2*DINNER]; xb = cols [0,512), gate = cols [512,1024).

__global__ void conv_silu_kernel(const float* __restrict__ inproj,
                                 const float* __restrict__ cw,
                                 const float* __restrict__ cb,
                                 float* __restrict__ xc32,
                                 f16* __restrict__ xc16) {
    int idx = blockIdx.x * blockDim.x + threadIdx.x;      // MROWS * DINNER
    if (idx >= MROWS * DINNER) return;
    int r = idx >> 9, c = idx & (DINNER - 1);
    int b = r >> 2, l = r & 3;
    float acc = cb[c];
    #pragma unroll
    for (int k = 0; k < DCONV; ++k) {
        int li = l - (DCONV - 1) + k;
        if (li >= 0)
            acc += cw[c * DCONV + k] * inproj[(size_t)(b * 4 + li) * (2 * DINNER) + c];
    }
    float v = silu_f(acc);
    xc32[idx] = v;
    xc16[idx] = (f16)v;
}

// ----------------------------- selective scan --------------------------------
// One thread per (b, d). 16-state recurrence over L=4, fused gate + f16 store.
// BC buffer holds [rows, 32]: cols 0..15 = B-proj, cols 16..31 = C-proj.

__global__ void ssm_scan_kernel(const float* __restrict__ dt,
                                const float* __restrict__ xc,
                                const float* __restrict__ BC,
                                const float* __restrict__ A_log,
                                const float* __restrict__ Dp,
                                const float* __restrict__ inproj,
                                f16* __restrict__ yg) {
    int idx = blockIdx.x * blockDim.x + threadIdx.x;      // B_SZ * DINNER
    if (idx >= B_SZ * DINNER) return;
    int b = idx >> 9, d = idx & (DINNER - 1);
    float a[DSTATE], h[DSTATE];
    #pragma unroll
    for (int n = 0; n < DSTATE; ++n) {
        a[n] = -__expf(A_log[(size_t)d * DSTATE + n]);
        h[n] = 0.f;
    }
    const float Dv = Dp[d];
    #pragma unroll
    for (int l = 0; l < NFRAMES; ++l) {
        int r = b * 4 + l;
        float dtv = dt[(size_t)r * DINNER + d];
        float xv  = xc[(size_t)r * DINNER + d];
        float dtx = dtv * xv;
        float y = 0.f;
        const float* bc = BC + (size_t)r * 32;
        #pragma unroll
        for (int n = 0; n < DSTATE; ++n) {
            float dA = __expf(dtv * a[n]);
            h[n] = dA * h[n] + dtx * bc[n];
            y += h[n] * bc[16 + n];
        }
        y += Dv * xv;
        float g = inproj[(size_t)r * (2 * DINNER) + DINNER + d];
        yg[(size_t)r * DINNER + d] = (f16)(y * silu_f(g));
    }
}

// ------------------------------ host launchers -------------------------------

static void launch_gemm(hipStream_t s, const f16* A, const f16* Bt, const float* bias,
                        float* C, f16* C16, const float* residual,
                        int M, int N, int Kp, int ldc, int epi) {
    int tilesN = (N + 31) / 32;
    int total  = (M / 32) * tilesN;
    int blocks = (total + 3) / 4;
    gemm_wmma_kernel<<<blocks, 128, 0, s>>>(A, Bt, bias, C, C16, residual,
                                            M, N, Kp, ldc, epi, tilesN, total);
}

static void launch_wcvt(hipStream_t s, const float* W, f16* Wt,
                        int K, int N, int Kp, int Np) {
    int total = Kp * Np;
    wcvt_kernel<<<(total + 255) / 256, 256, 0, s>>>(W, Wt, K, N, Kp, Np);
}

extern "C" void kernel_launch(void* const* d_in, const int* in_sizes, int n_in,
                              void* d_out, int out_size, void* d_ws, size_t ws_size,
                              hipStream_t stream) {
    (void)in_sizes; (void)n_in; (void)out_size; (void)ws_size;

    const float* obs        = (const float*)d_in[0];
    const float* embed_W    = (const float*)d_in[1];
    const float* embed_b    = (const float*)d_in[2];
    const float* embed_g    = (const float*)d_in[3];
    const float* embed_beta = (const float*)d_in[4];
    const float* pos_enc    = (const float*)d_in[5];
    const float* ln_g       = (const float*)d_in[6];
    const float* ln_b       = (const float*)d_in[7];
    const float* in_W       = (const float*)d_in[8];
    const float* conv_W     = (const float*)d_in[9];
    const float* conv_b     = (const float*)d_in[10];
    const float* dt_W       = (const float*)d_in[11];
    const float* dt_b       = (const float*)d_in[12];
    const float* B_W        = (const float*)d_in[13];
    const float* C_W        = (const float*)d_in[14];
    const float* A_log      = (const float*)d_in[15];
    const float* D_p        = (const float*)d_in[16];
    const float* out_W      = (const float*)d_in[17];
    const float* pool_g     = (const float*)d_in[18];
    const float* pool_beta  = (const float*)d_in[19];
    const float* pool_W     = (const float*)d_in[20];
    const float* pool_b     = (const float*)d_in[21];
    const float* p1_W       = (const float*)d_in[22];
    const float* p1_b       = (const float*)d_in[23];
    const float* p2_W       = (const float*)d_in[24];
    const float* p2_b       = (const float*)d_in[25];
    const float* ph_W       = (const float*)d_in[26];
    const float* ph_b       = (const float*)d_in[27];
    const float* v1_W       = (const float*)d_in[28];
    const float* v1_b       = (const float*)d_in[29];
    const float* v2_W       = (const float*)d_in[30];
    const float* v2_b       = (const float*)d_in[31];
    const float* vh_W       = (const float*)d_in[32];
    const float* vh_b       = (const float*)d_in[33];

    // ---- workspace bump allocator (256B aligned) ----
    char* base = (char*)d_ws;
    size_t off = 0;
    auto alloc = [&](size_t bytes) -> void* {
        void* p = base + off;
        off = (off + bytes + 255) & ~(size_t)255;
        return p;
    };

    // f16 transposed weights (Npad multiple of 32)
    f16* wt_embed = (f16*)alloc((size_t)256 * 128 * 2);
    f16* wt_in[NLAYERS]; f16* wt_dt[NLAYERS]; f16* wt_BC[NLAYERS]; f16* wt_out[NLAYERS];
    for (int i = 0; i < NLAYERS; ++i) {
        wt_in[i]  = (f16*)alloc((size_t)1024 * 256 * 2);
        wt_dt[i]  = (f16*)alloc((size_t)512 * 512 * 2);
        wt_BC[i]  = (f16*)alloc((size_t)32 * 512 * 2);   // rows 0-15: B_W^T, 16-31: C_W^T
        wt_out[i] = (f16*)alloc((size_t)256 * 512 * 2);
    }
    f16* wt_pool = (f16*)alloc((size_t)256 * 256 * 2);
    f16* wt_p1   = (f16*)alloc((size_t)256 * 256 * 2);
    f16* wt_p2   = (f16*)alloc((size_t)128 * 256 * 2);
    f16* wt_ph   = (f16*)alloc((size_t)32  * 128 * 2);
    f16* wt_v1   = (f16*)alloc((size_t)128 * 256 * 2);
    f16* wt_v2   = (f16*)alloc((size_t)64  * 128 * 2);
    f16* wt_vh   = (f16*)alloc((size_t)32  * 64  * 2);

    // activations
    f16*   A0      = (f16*)  alloc((size_t)MROWS * 128 * 2);
    float* ebuf    = (float*)alloc((size_t)MROWS * DMODEL * 4);
    float* x       = (float*)alloc((size_t)MROWS * DMODEL * 4);
    f16*   h16     = (f16*)  alloc((size_t)MROWS * DMODEL * 2);
    float* inproj  = (float*)alloc((size_t)MROWS * 2 * DINNER * 4);
    float* xc32    = (float*)alloc((size_t)MROWS * DINNER * 4);
    f16*   xc16    = (f16*)  alloc((size_t)MROWS * DINNER * 2);
    float* dtbuf   = (float*)alloc((size_t)MROWS * DINNER * 4);
    float* BCbuf   = (float*)alloc((size_t)MROWS * 32 * 4);
    f16*   yg16    = (f16*)  alloc((size_t)MROWS * DINNER * 2);
    f16*   lnf16   = (f16*)  alloc((size_t)B_SZ * DMODEL * 2);
    float* fscr    = (float*)alloc((size_t)B_SZ * DMODEL * 4);
    f16*   fh16    = (f16*)  alloc((size_t)B_SZ * DMODEL * 2);
    float* p1scr   = (float*)alloc((size_t)B_SZ * DMODEL * 4);
    f16*   p1h     = (f16*)  alloc((size_t)B_SZ * DMODEL * 2);
    float* p2scr   = (float*)alloc((size_t)B_SZ * 128 * 4);
    f16*   p2h     = (f16*)  alloc((size_t)B_SZ * 128 * 2);
    float* v1scr   = (float*)alloc((size_t)B_SZ * 128 * 4);
    f16*   v1h     = (f16*)  alloc((size_t)B_SZ * 128 * 2);
    float* v2scr   = (float*)alloc((size_t)B_SZ * 64 * 4);
    f16*   v2h     = (f16*)  alloc((size_t)B_SZ * 64 * 2);

    // ---- weight conversion (transposed + padded f16) ----
    launch_wcvt(stream, embed_W, wt_embed, FRAME, DMODEL, 128, 256);
    for (int i = 0; i < NLAYERS; ++i) {
        launch_wcvt(stream, in_W  + (size_t)i * 256 * 1024, wt_in[i],  256, 1024, 256, 1024);
        launch_wcvt(stream, dt_W  + (size_t)i * 512 * 512,  wt_dt[i],  512, 512,  512, 512);
        launch_wcvt(stream, B_W   + (size_t)i * 512 * 16,   wt_BC[i],            512, 16, 512, 16);
        launch_wcvt(stream, C_W   + (size_t)i * 512 * 16,   wt_BC[i] + 16 * 512, 512, 16, 512, 16);
        launch_wcvt(stream, out_W + (size_t)i * 512 * 256,  wt_out[i], 512, 256,  512, 256);
    }
    launch_wcvt(stream, pool_W, wt_pool, 256, 256, 256, 256);
    launch_wcvt(stream, p1_W,   wt_p1,   256, 256, 256, 256);
    launch_wcvt(stream, p2_W,   wt_p2,   256, 128, 256, 128);
    launch_wcvt(stream, ph_W,   wt_ph,   128, 19,  128, 32);
    launch_wcvt(stream, v1_W,   wt_v1,   256, 128, 256, 128);
    launch_wcvt(stream, v2_W,   wt_v2,   128, 64,  128, 64);
    launch_wcvt(stream, vh_W,   wt_vh,   64,  1,   64,  32);

    // ---- embedding ----
    obs_pad_kernel<<<(MROWS * 128) / 256, 256, 0, stream>>>(obs, A0);
    launch_gemm(stream, A0, wt_embed, embed_b, ebuf, nullptr, nullptr,
                MROWS, DMODEL, 128, DMODEL, EPI_NONE);
    embed_post_kernel<<<MROWS / 8, 256, 0, stream>>>(ebuf, embed_g, embed_beta, pos_enc, x);

    // ---- mamba layers ----
    for (int i = 0; i < NLAYERS; ++i) {
        preln_kernel<<<MROWS / 8, 256, 0, stream>>>(x, ln_g + i * DMODEL, ln_b + i * DMODEL, h16);
        launch_gemm(stream, h16, wt_in[i], nullptr, inproj, nullptr, nullptr,
                    MROWS, 2 * DINNER, 256, 2 * DINNER, EPI_NONE);
        conv_silu_kernel<<<(MROWS * DINNER) / 256, 256, 0, stream>>>(
            inproj, conv_W + (size_t)i * DINNER * DCONV, conv_b + (size_t)i * DINNER,
            xc32, xc16);
        launch_gemm(stream, xc16, wt_dt[i], dt_b + (size_t)i * DINNER, dtbuf, nullptr, nullptr,
                    MROWS, DINNER, 512, DINNER, EPI_SOFTPLUS);
        launch_gemm(stream, xc16, wt_BC[i], nullptr, BCbuf, nullptr, nullptr,
                    MROWS, 32, 512, 32, EPI_NONE);
        ssm_scan_kernel<<<(B_SZ * DINNER) / 256, 256, 0, stream>>>(
            dtbuf, xc32, BCbuf, A_log + (size_t)i * DINNER * DSTATE,
            D_p + (size_t)i * DINNER, inproj, yg16);
        launch_gemm(stream, yg16, wt_out[i], nullptr, x, nullptr, /*residual=*/x,
                    MROWS, DMODEL, 512, DMODEL, EPI_NONE);
    }

    // ---- pool + heads ----
    pool_ln_kernel<<<B_SZ / 8, 256, 0, stream>>>(x, pool_g, pool_beta, lnf16);
    launch_gemm(stream, lnf16, wt_pool, pool_b, fscr, fh16, nullptr,
                B_SZ, DMODEL, 256, DMODEL, EPI_MISH);
    // policy
    launch_gemm(stream, fh16, wt_p1, p1_b, p1scr, p1h, nullptr,
                B_SZ, DMODEL, 256, DMODEL, EPI_MISH);
    launch_gemm(stream, p1h, wt_p2, p2_b, p2scr, p2h, nullptr,
                B_SZ, 128, 256, 128, EPI_MISH);
    launch_gemm(stream, p2h, wt_ph, ph_b, (float*)d_out, nullptr, nullptr,
                B_SZ, 19, 128, 20, EPI_NONE);
    // value
    launch_gemm(stream, fh16, wt_v1, v1_b, v1scr, v1h, nullptr,
                B_SZ, 128, 256, 128, EPI_MISH);
    launch_gemm(stream, v1h, wt_v2, v2_b, v2scr, v2h, nullptr,
                B_SZ, 64, 128, 64, EPI_MISH);
    launch_gemm(stream, v2h, wt_vh, vh_b, (float*)d_out + 19, nullptr, nullptr,
                B_SZ, 1, 64, 20, EPI_NONE);
}
